// CoGSL_32985348833480
// MI455X (gfx1250) — matrix operand order repaired and checked by reading, not compiled, not run on hardware
//
#include <hip/hip_runtime.h>

// ---------------- problem constants (match reference) ----------------
#define Nn   30000
#define Ee   480000
#define Fd   256
#define GH   64      // GEN_HID
#define CH   128     // CLS_HID
#define NC   16      // NUM_CLASS
#define MH   128     // MI_HID
#define BT   4096    // BATCH
#define EPSc 1e-8f
#define TAUc 0.5f

typedef __attribute__((ext_vector_type(16))) _Float16 v16h;
typedef __attribute__((ext_vector_type(8)))  float    v8f;

// ---------------------------------------------------------------------
// WMMA GEMM:  C[M x N] = A[M x K] @ B[K x N]   (fp32 in/out, f16 WMMA)
// One wave computes a 16 x (16*NT) strip; K multiple of 32.
// Fragment layouts per CDNA5 ISA 7.12.2 (wave32):
//   A 16x32 f16 : lane = {m=lane&15, kh=lane>>4}; elem e -> k =
//                 (e&8?16:0) + (e&6) + (e&1) + kh*8
//   B 32x16 f16 : lane = {n=lane&15, kh=lane>>4}; elem e -> k = (e&8?16:0) + kh*8 + (e&7)
//   C 16x16 f32 : vgpr v, lanes 0-15 -> (row v,   col lane)
//                           lanes16-31 -> (row v+8, col lane-16)
// act: 0 = none, 1 = ELU(alpha=1)
// ---------------------------------------------------------------------
template<int NT>
__global__ void gemm_wmma_kernel(const float* __restrict__ A,
                                 const float* __restrict__ Bm,
                                 const float* __restrict__ bias,
                                 float* __restrict__ C,
                                 int M, int K, int N, int act)
{
    const int lane = threadIdx.x & 31;
    const int wave = (blockIdx.x * blockDim.x + threadIdx.x) >> 5;
    const int mtiles  = (M + 15) >> 4;
    const int nstrips = N / (16 * NT);
    if (wave >= mtiles * nstrips) return;
    const int mt  = wave / nstrips;
    const int nt0 = (wave - mt * nstrips) * (16 * NT);

    const int m  = lane & 15;
    const int kh = lane >> 4;
    const int row = mt * 16 + m;
    const bool rok = (row < M);

    v8f acc[NT];
#pragma unroll
    for (int t = 0; t < NT; ++t)
#pragma unroll
        for (int q = 0; q < 8; ++q) acc[t][q] = 0.f;

    for (int k0 = 0; k0 < K; k0 += 32) {
        v16h a;
#pragma unroll
        for (int e = 0; e < 16; ++e) {
            int v = e >> 1, p = e & 1;
            int k = ((v & 4) ? 16 : 0) + (v & 3) * 2 + p + kh * 8;
            a[e] = rok ? (_Float16)A[(size_t)row * K + k0 + k] : (_Float16)0.f;
        }
#pragma unroll
        for (int t = 0; t < NT; ++t) {
            const int col = nt0 + t * 16 + m;
            v16h b;
#pragma unroll
            for (int e = 0; e < 16; ++e) {
                int k = ((e & 8) ? 16 : 0) + kh * 8 + (e & 7);
                b[e] = (_Float16)Bm[(size_t)(k0 + k) * N + col];
            }
            acc[t] = __builtin_amdgcn_wmma_f32_16x16x32_f16(
                false, a, false, b, (short)0, acc[t], false, false);
        }
    }

#pragma unroll
    for (int t = 0; t < NT; ++t) {
#pragma unroll
        for (int v = 0; v < 8; ++v) {
            int r = mt * 16 + v + kh * 8;
            int c = nt0 + t * 16 + m;
            if (r < M) {
                float val = acc[t][v];
                if (bias) val += bias[c];
                if (act == 1) val = val > 0.f ? val : (expf(val) - 1.f);
                C[(size_t)r * N + c] = val;
            }
        }
    }
}

// ---------------- elementwise helpers ----------------
__global__ void fill_kernel(float* p, float v, int n)
{
    int i = blockIdx.x * blockDim.x + threadIdx.x;
    if (i < n) p[i] = v;
}

__global__ void relu_kernel(float* p, int n)
{
    int i = blockIdx.x * blockDim.x + threadIdx.x;
    if (i < n) p[i] = fmaxf(p[i], 0.f);
}

// ---------------- SPMM: Y[row] += (vals*scale[row]) * X[col]  ----------------
// float4 gathers + scalar f32 atomic scatters (bandwidth-bound path).
__global__ void spmm_edges_kernel(const int* __restrict__ r, const int* __restrict__ c,
                                  const float* __restrict__ vals,
                                  const float* __restrict__ scale,
                                  const float* __restrict__ X, float* __restrict__ Y,
                                  int E, int F)
{
    const int chunks = F >> 2;
    int t = blockIdx.x * blockDim.x + threadIdx.x;
    if (t >= E * chunks) return;
    int e = t / chunks;
    int f = (t - e * chunks) << 2;
    int row = r[e], col = c[e];
    float v = vals[e];
    if (scale) v *= scale[row];
    const float4 x = *reinterpret_cast<const float4*>(X + (size_t)col * F + f);
    float* y = Y + (size_t)row * F + f;
    atomicAdd(y + 0, v * x.x);
    atomicAdd(y + 1, v * x.y);
    atomicAdd(y + 2, v * x.z);
    atomicAdd(y + 3, v * x.w);
}

// ---------------- edge attention score ----------------
__global__ void edge_score_kernel(const int* __restrict__ r, const int* __restrict__ c,
                                  const float* __restrict__ emb,
                                  const float* __restrict__ Wm, const float* __restrict__ bm,
                                  float* __restrict__ score, int E, int G)
{
    int e = blockIdx.x * blockDim.x + threadIdx.x;
    if (e >= E) return;
    const float* ei = emb + (size_t)r[e] * G;
    const float* ej = emb + (size_t)c[e] * G;
    float s = bm[0];
    for (int g = 0; g < G; ++g) s += ei[g] * Wm[g] + ej[g] * Wm[G + g];
    score[e] = s;
}

// ---------------- sparse row softmax pieces ----------------
__device__ inline void atomicMaxF(float* addr, float val)
{
    unsigned int* ua = (unsigned int*)addr;
    unsigned int old = __float_as_uint(*addr);
    while (__uint_as_float(old) < val) {
        unsigned int assumed = old;
        old = atomicCAS(ua, assumed, __float_as_uint(val));
        if (old == assumed) break;
    }
}

__global__ void segmax_kernel(const int* __restrict__ rows, const float* __restrict__ score,
                              float* __restrict__ m, int E)
{
    int e = blockIdx.x * blockDim.x + threadIdx.x;
    if (e < E) atomicMaxF(&m[rows[e]], score[e]);
}

__global__ void segexp_kernel(const int* __restrict__ rows, float* __restrict__ score,
                              const float* __restrict__ m, float* __restrict__ ssum, int E)
{
    int e = blockIdx.x * blockDim.x + threadIdx.x;
    if (e >= E) return;
    int r = rows[e];
    float ev = expf(score[e] - m[r]);
    score[e] = ev;
    atomicAdd(&ssum[r], ev);
}

__global__ void newvals_kernel(const int* __restrict__ rows, const float* __restrict__ vals,
                               const float* __restrict__ ev, const float* __restrict__ ssum,
                               float lam, float* __restrict__ nv, int E)
{
    int e = blockIdx.x * blockDim.x + threadIdx.x;
    if (e >= E) return;
    nv[e] = vals[e] + lam * ev[e] / ssum[rows[e]];
}

// ---------------- dense 16-wide softmax (+ optional prob out, logits out) ----------------
__global__ void softmax16_kernel(const float* __restrict__ in, float* __restrict__ prob,
                                 float* __restrict__ logits, int n)
{
    int i = blockIdx.x * blockDim.x + threadIdx.x;
    if (i >= n) return;
    float v[16]; float mx = -3.0e38f;
    for (int c2 = 0; c2 < 16; ++c2) { v[c2] = in[(size_t)i * 16 + c2]; mx = fmaxf(mx, v[c2]); }
    float s = 0.f;
    for (int c2 = 0; c2 < 16; ++c2) { v[c2] = expf(v[c2] - mx); s += v[c2]; }
    float inv = 1.f / s;
    for (int c2 = 0; c2 < 16; ++c2) {
        float p = v[c2] * inv;
        if (prob) prob[(size_t)i * 16 + c2] = p;
        logits[(size_t)i * 16 + c2] = logf(p + EPSc);
    }
}

// ---------------- get_weight + b1/b2 ----------------
__global__ void weight_b_kernel(const float* __restrict__ p1, const float* __restrict__ p2,
                                float* __restrict__ b1, float* __restrict__ b2, int n)
{
    int i = blockIdx.x * blockDim.x + threadIdx.x;
    if (i >= n) return;
    auto topw = [](const float* p) -> float {
        float f = -1e30f, s2 = -1e30f;
        for (int c = 0; c < 16; ++c) {
            float x = p[c];
            if (x > f) { s2 = f; f = x; }
            else if (x > s2) s2 = x;
        }
        // ALPHA=1, LAM=0.5
        return expf(0.5f * logf(f + EPSc) + 0.5f * logf(f - s2 + EPSc));
    };
    float w1 = topw(p1 + (size_t)i * 16);
    float w2 = topw(p2 + (size_t)i * 16);
    float inv = 1.f / (w1 + w2);
    b1[i] = w1 * inv;
    b2[i] = w2 * inv;
}

// ---------------- gather / row norm ----------------
__global__ void gather_kernel(const float* __restrict__ in, const int* __restrict__ idx,
                              float* __restrict__ outp, int B, int F)
{
    int t = blockIdx.x * blockDim.x + threadIdx.x;
    if (t >= B * F) return;
    int i = t / F, f = t - i * F;
    outp[t] = in[(size_t)idx[i] * F + f];
}

__global__ void rownorm_kernel(const float* __restrict__ z, float* __restrict__ nrm, int B, int F)
{
    int i = blockIdx.x * blockDim.x + threadIdx.x;
    if (i >= B) return;
    float s = 0.f;
    for (int f = 0; f < F; ++f) { float x = z[(size_t)i * F + f]; s += x * x; }
    nrm[i] = sqrtf(s);
}

// ---------------- fused contrastive tile kernel ----------------
// Wave computes 16x16 tile of exp(z1@z2^T / (n1 n2^T) / TAU) via WMMA and
// reduces it into rowsum / colsum / diag without materializing 4096x4096.
__global__ void contrast_tiles_kernel(const float* __restrict__ z1, const float* __restrict__ z2,
                                      const float* __restrict__ n1, const float* __restrict__ n2,
                                      float* __restrict__ rowsum, float* __restrict__ colsum,
                                      float* __restrict__ diag)
{
    const int lane = threadIdx.x & 31;
    const int wave = (blockIdx.x * blockDim.x + threadIdx.x) >> 5;
    const int T = BT / 16;
    if (wave >= T * T) return;
    const int it = wave / T, jt = wave - it * T;
    const int m = lane & 15, kh = lane >> 4;
    const int gi0 = it * 16;
    const int gj  = jt * 16 + m;
    const int rowA = gi0 + m;

    v8f acc;
#pragma unroll
    for (int q = 0; q < 8; ++q) acc[q] = 0.f;

    for (int k0 = 0; k0 < MH; k0 += 32) {
        v16h a, b;
#pragma unroll
        for (int e = 0; e < 16; ++e) {
            int v = e >> 1, p = e & 1;
            int k = ((v & 4) ? 16 : 0) + (v & 3) * 2 + p + kh * 8;
            a[e] = (_Float16)z1[(size_t)rowA * MH + k0 + k];
        }
#pragma unroll
        for (int e = 0; e < 16; ++e) {
            int k = ((e & 8) ? 16 : 0) + kh * 8 + (e & 7);   // B[k][n] = z2[n][k]
            b[e] = (_Float16)z2[(size_t)gj * MH + k0 + k];
        }
        acc = __builtin_amdgcn_wmma_f32_16x16x32_f16(
            false, a, false, b, (short)0, acc, false, false);
    }

    const float n2j = n2[gj];
    float colpart = 0.f;
#pragma unroll
    for (int v = 0; v < 8; ++v) {
        int gi = gi0 + v + kh * 8;
        float ev = expf(acc[v] / (n1[gi] * n2j * TAUc));
        if (gi == gj) diag[gi] = ev;
        float rsum = ev;
        rsum += __shfl_xor(rsum, 1, 32);
        rsum += __shfl_xor(rsum, 2, 32);
        rsum += __shfl_xor(rsum, 4, 32);
        rsum += __shfl_xor(rsum, 8, 32);
        if (m == 0) atomicAdd(&rowsum[gi], rsum);
        colpart += ev;
    }
    colpart += __shfl_xor(colpart, 16, 32);
    if (kh == 0) atomicAdd(&colsum[gj], colpart);
}

__global__ void contrast_finalize_kernel(const float* __restrict__ rs, const float* __restrict__ cs,
                                         const float* __restrict__ dg, float* __restrict__ o)
{
    __shared__ float sa[256], sb[256];
    float a = 0.f, b = 0.f;
    for (int i = threadIdx.x; i < BT; i += 256) {
        float d = dg[i];
        a += -logf(d / (rs[i] + EPSc) + EPSc);
        b += -logf(d / (cs[i] + EPSc) + EPSc);
    }
    sa[threadIdx.x] = a; sb[threadIdx.x] = b;
    __syncthreads();
    for (int s = 128; s > 0; s >>= 1) {
        if ((int)threadIdx.x < s) { sa[threadIdx.x] += sa[threadIdx.x + s]; sb[threadIdx.x] += sb[threadIdx.x + s]; }
        __syncthreads();
    }
    if (threadIdx.x == 0) *o = 0.5f * (sa[0] + sb[0]) / (float)BT;
}

// ---------------- host-side launch helpers ----------------
template<int NT>
static void launch_gemm(const float* A, const float* B, const float* bias, float* C,
                        int M, int K, int N, int act, hipStream_t s)
{
    int waves  = ((M + 15) / 16) * (N / (16 * NT));
    int blocks = (waves + 7) / 8;                 // 8 waves (256 thr) per block
    gemm_wmma_kernel<NT><<<blocks, 256, 0, s>>>(A, B, bias, C, M, K, N, act);
}

static void fillf(float* p, float v, int n, hipStream_t s)
{
    fill_kernel<<<(n + 255) / 256, 256, 0, s>>>(p, v, n);
}

static void reluf(float* p, int n, hipStream_t s)
{
    relu_kernel<<<(n + 255) / 256, 256, 0, s>>>(p, n);
}

static void spmm_launch(const int* r, const int* c, const float* vals, const float* scale,
                        const float* X, float* Y, int E, int F, hipStream_t s)
{
    int total  = E * (F >> 2);
    int blocks = (total + 255) / 256;
    spmm_edges_kernel<<<blocks, 256, 0, s>>>(r, c, vals, scale, X, Y, E, F);
}

// ---------------------------------------------------------------------
extern "C" void kernel_launch(void* const* d_in, const int* in_sizes, int n_in,
                              void* d_out, int out_size, void* d_ws, size_t ws_size,
                              hipStream_t stream)
{
    (void)in_sizes; (void)n_in; (void)out_size; (void)ws_size;

    const float* feat  = (const float*)d_in[0];
    const int*   v1r   = (const int*)d_in[1];
    const int*   v1c   = v1r + Ee;
    const float* v1v   = (const float*)d_in[2];
    const int*   v2r   = (const int*)d_in[3];
    const int*   v2c   = v2r + Ee;
    const float* v2v   = (const float*)d_in[4];
    const int*   idx   = (const int*)d_in[5];
    const float* Wgen1 = (const float*)d_in[6];
    const float* Wm1   = (const float*)d_in[7];
    const float* bm1   = (const float*)d_in[8];
    const float* Wgen2 = (const float*)d_in[9];
    const float* Wm2   = (const float*)d_in[10];
    const float* bm2   = (const float*)d_in[11];
    const float* W1v1  = (const float*)d_in[12];
    const float* W2v1  = (const float*)d_in[13];
    const float* W1v2  = (const float*)d_in[14];
    const float* W2v2  = (const float*)d_in[15];
    const float* W1v   = (const float*)d_in[16];
    const float* W2v   = (const float*)d_in[17];
    const float* Wg    = (const float*)d_in[18];
    const float* Wg1   = (const float*)d_in[19];
    const float* Wg2   = (const float*)d_in[20];
    const float* Wp1   = (const float*)d_in[21];
    const float* bp1   = (const float*)d_in[22];
    const float* Wp2   = (const float*)d_in[23];
    const float* bp2   = (const float*)d_in[24];
    float* out = (float*)d_out;

    // --------- workspace layout (~52 MB, all re-initialized each call) ---------
    float* ws  = (float*)d_ws;
    float* A   = ws; ws += (size_t)Nn * CH;   // GEMM results (widest use: 30000x128)
    float* Bb  = ws; ws += (size_t)Nn * CH;   // SPMM results / hidden
    float* SC  = ws; ws += Ee;                // edge scores / 30000x16 spmm buffer
    float* SM  = ws; ws += Nn;                // segment max
    float* SS  = ws; ws += Nn;                // segment sum
    float* NV1 = ws; ws += Ee;                // new view-1 values
    float* NV2 = ws; ws += Ee;                // new view-2 values
    float* PB1 = ws; ws += (size_t)Nn * NC;   // prob_v1
    float* PB2 = ws; ws += (size_t)Nn * NC;   // prob_v2
    float* B1  = ws; ws += Nn;                // b1
    float* B2  = ws; ws += Nn;                // b2
    float* GA  = ws; ws += (size_t)BT * MH;   // gathered emb
    float* GB  = ws; ws += (size_t)BT * MH;   // proj hidden
    float* EV  = ws; ws += (size_t)BT * MH;   // e_v
    float* E1  = ws; ws += (size_t)BT * MH;   // e_1
    float* E2  = ws; ws += (size_t)BT * MH;   // e_2
    float* NRv = ws; ws += BT;
    float* NR1 = ws; ws += BT;
    float* NR2 = ws; ws += BT;
    float* RS  = ws; ws += BT;                // contrast row sums
    float* CS  = ws; ws += BT;                // contrast col sums
    float* DG  = ws; ws += BT;                // contrast diag

    // ================= gen_view (original, unsymmetrized graphs) =================
    auto gen_view = [&](const int* r, const int* c, const float* vv,
                        const float* Wgen, const float* Wm, const float* bm, float* nv) {
        launch_gemm<4>(feat, Wgen, nullptr, A, Nn, Fd, GH, 0, stream);
        fillf(Bb, 0.f, Nn * GH, stream);
        spmm_launch(r, c, vv, nullptr, A, Bb, Ee, GH, stream);
        reluf(Bb, Nn * GH, stream);
        edge_score_kernel<<<(Ee + 255) / 256, 256, 0, stream>>>(r, c, Bb, Wm, bm, SC, Ee, GH);
        fillf(SM, -3.0e38f, Nn, stream);
        segmax_kernel<<<(Ee + 255) / 256, 256, 0, stream>>>(r, SC, SM, Ee);
        fillf(SS, 0.f, Nn, stream);
        segexp_kernel<<<(Ee + 255) / 256, 256, 0, stream>>>(r, SC, SM, SS, Ee);
        newvals_kernel<<<(Ee + 255) / 256, 256, 0, stream>>>(r, vv, SC, SS, 0.5f, nv, Ee);
    };
    gen_view(v1r, v1c, v1v, Wgen1, Wm1, bm1, NV1);
    gen_view(v2r, v2c, v2v, Wgen2, Wm2, bm2, NV2);

    // ================= per-view classifiers (symmetrized graphs) =================
    auto classify = [&](const int* r, const int* c, const float* nv,
                        const float* W1, const float* W2, float* prob, float* logits) {
        launch_gemm<8>(feat, W1, nullptr, A, Nn, Fd, CH, 0, stream);
        fillf(Bb, 0.f, Nn * CH, stream);
        spmm_launch(r, c, nv, nullptr, A, Bb, Ee, CH, stream);   // forward edges
        spmm_launch(c, r, nv, nullptr, A, Bb, Ee, CH, stream);   // reversed edges (sym)
        reluf(Bb, Nn * CH, stream);
        launch_gemm<1>(Bb, W2, nullptr, A, Nn, CH, NC, 0, stream);
        fillf(SC, 0.f, Nn * NC, stream);
        spmm_launch(r, c, nv, nullptr, A, SC, Ee, NC, stream);
        spmm_launch(c, r, nv, nullptr, A, SC, Ee, NC, stream);
        softmax16_kernel<<<(Nn + 255) / 256, 256, 0, stream>>>(SC, prob, logits, Nn);
    };
    classify(v1r, v1c, NV1, W1v1, W2v1, PB1, out + (size_t)Nn * NC);
    classify(v2r, v2c, NV2, W1v2, W2v2, PB2, out + 2 * (size_t)Nn * NC);

    weight_b_kernel<<<(Nn + 255) / 256, 256, 0, stream>>>(PB1, PB2, B1, B2, Nn);

    // ================= fused-view classifier =================
    launch_gemm<8>(feat, W1v, nullptr, A, Nn, Fd, CH, 0, stream);
    fillf(Bb, 0.f, Nn * CH, stream);
    spmm_launch(v1r, v1c, NV1, B1, A, Bb, Ee, CH, stream);
    spmm_launch(v1c, v1r, NV1, B1, A, Bb, Ee, CH, stream);
    spmm_launch(v2r, v2c, NV2, B2, A, Bb, Ee, CH, stream);
    spmm_launch(v2c, v2r, NV2, B2, A, Bb, Ee, CH, stream);
    reluf(Bb, Nn * CH, stream);
    launch_gemm<1>(Bb, W2v, nullptr, A, Nn, CH, NC, 0, stream);
    fillf(SC, 0.f, Nn * NC, stream);
    spmm_launch(v1r, v1c, NV1, B1, A, SC, Ee, NC, stream);
    spmm_launch(v1c, v1r, NV1, B1, A, SC, Ee, NC, stream);
    spmm_launch(v2r, v2c, NV2, B2, A, SC, Ee, NC, stream);
    spmm_launch(v2c, v2r, NV2, B2, A, SC, Ee, NC, stream);
    softmax16_kernel<<<(Nn + 255) / 256, 256, 0, stream>>>(SC, nullptr, out, Nn);

    // ================= MI encoders + projection (gather-then-project) =================
    auto mi_branch = [&](const float* Wg_, int mode, float* Eout) {
        launch_gemm<8>(feat, Wg_, nullptr, A, Nn, Fd, MH, 0, stream);
        fillf(Bb, 0.f, Nn * MH, stream);
        if (mode == 0) {            // fused graph with b1/b2 row scales
            spmm_launch(v1r, v1c, NV1, B1, A, Bb, Ee, MH, stream);
            spmm_launch(v1c, v1r, NV1, B1, A, Bb, Ee, MH, stream);
            spmm_launch(v2r, v2c, NV2, B2, A, Bb, Ee, MH, stream);
            spmm_launch(v2c, v2r, NV2, B2, A, Bb, Ee, MH, stream);
        } else if (mode == 1) {     // sym view 1
            spmm_launch(v1r, v1c, NV1, nullptr, A, Bb, Ee, MH, stream);
            spmm_launch(v1c, v1r, NV1, nullptr, A, Bb, Ee, MH, stream);
        } else {                    // sym view 2
            spmm_launch(v2r, v2c, NV2, nullptr, A, Bb, Ee, MH, stream);
            spmm_launch(v2c, v2r, NV2, nullptr, A, Bb, Ee, MH, stream);
        }
        reluf(Bb, Nn * MH, stream);
        gather_kernel<<<((BT * MH) + 255) / 256, 256, 0, stream>>>(Bb, idx, GA, BT, MH);
        launch_gemm<8>(GA, Wp1, bp1, GB, BT, MH, MH, 1, stream);   // ELU epilogue
        launch_gemm<8>(GB, Wp2, bp2, Eout, BT, MH, MH, 0, stream);
    };
    mi_branch(Wg,  0, EV);
    mi_branch(Wg1, 1, E1);
    mi_branch(Wg2, 2, E2);

    rownorm_kernel<<<(BT + 255) / 256, 256, 0, stream>>>(EV, NRv, BT, MH);
    rownorm_kernel<<<(BT + 255) / 256, 256, 0, stream>>>(E1, NR1, BT, MH);
    rownorm_kernel<<<(BT + 255) / 256, 256, 0, stream>>>(E2, NR2, BT, MH);

    // ================= fused contrastive losses =================
    float* scal = out + 3 * (size_t)Nn * NC;
    auto contrast = [&](const float* z1, const float* n1,
                        const float* z2, const float* n2, float* o) {
        fillf(RS, 0.f, BT, stream);
        fillf(CS, 0.f, BT, stream);
        fillf(DG, 0.f, BT, stream);
        int waves = (BT / 16) * (BT / 16);
        contrast_tiles_kernel<<<(waves + 7) / 8, 256, 0, stream>>>(z1, z2, n1, n2, RS, CS, DG);
        contrast_finalize_kernel<<<1, 256, 0, stream>>>(RS, CS, DG, o);
    };
    contrast(EV, NRv, E1, NR1, scal + 0);   // vv1
    contrast(EV, NRv, E2, NR2, scal + 1);   // vv2
    contrast(E1, NR1, E2, NR2, scal + 2);   // v1v2
}
